// RelAttention_15298673509083
// MI455X (gfx1250) — matrix operand order repaired and checked by reading, not compiled, not run on hardware
//
#include <hip/hip_runtime.h>
#include <hip/hip_bf16.h>

// ---------------------------------------------------------------------------
// Relative attention (Transformer-XL style) for MI455X (gfx1250).
// All matmuls via v_wmma_f32_16x16x32_bf16 (wave32 WMMA), scores kept in LDS.
// ---------------------------------------------------------------------------

#define S_LEN   1024
#define DMODEL  1024
#define NH      16
#define DH      64
#define BATCH   4
#define BHEADS  (BATCH * NH)          // 64
#define NTOK    (BATCH * S_LEN)       // 4096
#define ATT_SCALE 0.03125f            // 1/sqrt(DMODEL)

#define SROW    1032                  // score row stride (floats), padded
#define QSTRIDE 72                    // Q tile row stride (bf16 elems), padded

typedef __attribute__((ext_vector_type(16))) __bf16 v16bf;
typedef __attribute__((ext_vector_type(8)))  float  v8f;

union FragBits { uint4 q[2]; v16bf v; };

__device__ __forceinline__ v8f wmma_bf16f32(v16bf a, v16bf b, v8f c) {
    // D = A(16x32 bf16) * B(32x16 bf16) + C(16x16 f32)
    return __builtin_amdgcn_wmma_f32_16x16x32_bf16(
        false, a, false, b, (short)0, c, false, false);
}

// A-matrix fragment (16x32, bf16, row-major source).
// Lane l: M = l&15, K chunks at {half*8 .. +7} and {half*8+16 .. +23}.
__device__ __forceinline__ v16bf load_frag_a(const __bf16* base, int row0,
                                             int stride, int k0, int lane) {
    const int m = lane & 15, half = lane >> 4;
    const __bf16* p = base + (row0 + m) * stride + k0 + half * 8;
    FragBits f;
    f.q[0] = *(const uint4*)(p);
    f.q[1] = *(const uint4*)(p + 16);
    return f.v;
}

// B-matrix fragment (32x16, bf16). Column n = lane&15 comes from a row of the
// (transposed-operand) source, K chunks at {half*8 .. +7} and {half*8+16 .. +23}.
__device__ __forceinline__ v16bf load_frag_b(const __bf16* base, int col0,
                                             int stride, int k0, int lane) {
    const int n = lane & 15, half = lane >> 4;
    const __bf16* p = base + (col0 + n) * stride + k0 + half * 8;
    FragBits f;
    f.q[0] = *(const uint4*)(p);
    f.q[1] = *(const uint4*)(p + 16);
    return f.v;
}

// A fragment from an f32 LDS score tile, scaled by per-row 1/sum, cvt to bf16.
__device__ __forceinline__ v16bf load_frag_a_probs(const float* sc, int row0,
                                                   int stride, int k0,
                                                   float rs, int lane) {
    const int m = lane & 15, half = lane >> 4;
    const float* p = sc + (row0 + m) * stride + k0 + half * 8;
    v16bf out;
#pragma unroll
    for (int i = 0; i < 8; ++i) {
        out[i]     = (__bf16)(p[i] * rs);
        out[i + 8] = (__bf16)(p[i + 16] * rs);
    }
    return out;
}

// ---------------------------------------------------------------------------
// Kernel 0: f32 -> bf16 conversion (vectorized x4)
// ---------------------------------------------------------------------------
__global__ __launch_bounds__(256) void cvt_bf16_kernel(const float* __restrict__ in,
                                                       __bf16* __restrict__ out,
                                                       int n4) {
    int i = blockIdx.x * 256 + threadIdx.x;
    if (i < n4) {
        float4 f = ((const float4*)in)[i];
        out[i * 4 + 0] = (__bf16)f.x;
        out[i * 4 + 1] = (__bf16)f.y;
        out[i * 4 + 2] = (__bf16)f.z;
        out[i * 4 + 3] = (__bf16)f.w;
    }
}

// ---------------------------------------------------------------------------
// Kernel 1: C = A @ W^T + bias    (A: [4096,1024] bf16, W: [1024,1024] bf16)
// mode 0: bf16 row-major out   (Q, K projections)
// mode 1: bf16 out transposed per head -> Vt[bh][dh][s]   (V projection)
// mode 2: f32 row-major out    (final output projection)
// grid = (16 col-tiles, 64 row-tiles), 256 threads = 8 waves, 64x64 per block.
// ---------------------------------------------------------------------------
__global__ __launch_bounds__(256) void gemm_wmma_kernel(
    const __bf16* __restrict__ A, const __bf16* __restrict__ W,
    const float* __restrict__ bias, __bf16* __restrict__ outb,
    float* __restrict__ outf, int mode) {
    const int i0 = blockIdx.y * 64;
    const int f0 = blockIdx.x * 64;
    const int wave = threadIdx.x >> 5, lane = threadIdx.x & 31;

    // wave handles two 16x16 tiles sharing the same A row-block
    const int rb  = wave >> 1;
    const int m0  = i0 + rb * 16;
    const int n0  = f0 + (wave & 1) * 32;
    const int n1  = n0 + 16;

    v8f acc0 = {}, acc1 = {};
    for (int kk = 0; kk < DMODEL; kk += 32) {
        v16bf a  = load_frag_a(A, m0, DMODEL, kk, lane);
        v16bf b0 = load_frag_b(W, n0, DMODEL, kk, lane);
        v16bf b1 = load_frag_b(W, n1, DMODEL, kk, lane);
        acc0 = wmma_bf16f32(a, b0, acc0);
        acc1 = wmma_bf16f32(a, b1, acc1);
    }

    const int half = lane >> 4, nn = lane & 15;
#pragma unroll
    for (int r = 0; r < 8; ++r) {
        const int row = m0 + r + half * 8;
        const float v0 = acc0[r] + bias[n0 + nn];
        const float v1 = acc1[r] + bias[n1 + nn];
        if (mode == 0) {
            outb[row * DMODEL + n0 + nn] = (__bf16)v0;
            outb[row * DMODEL + n1 + nn] = (__bf16)v1;
        } else if (mode == 1) {
            const int b = row >> 10, s = row & (S_LEN - 1);
            int c = n0 + nn;
            outb[((b * NH + (c >> 6)) * DH + (c & 63)) * S_LEN + s] = (__bf16)v0;
            c = n1 + nn;
            outb[((b * NH + (c >> 6)) * DH + (c & 63)) * S_LEN + s] = (__bf16)v1;
        } else {
            outf[row * DMODEL + n0 + nn] = v0;
            outf[row * DMODEL + n1 + nn] = v1;
        }
    }
}

// ---------------------------------------------------------------------------
// Kernel 2: fused relative attention for one (head, 64-query tile).
//   sc = (q+u)K^T ; sc += rel_shift((q+v)R^T) ; softmax(sc*SCALE) ; O = sc@V
// Dynamic LDS: 64x1032 f32 scores + qu(64x72) + qv(80x72) bf16 + 64 f32 rinv.
// ---------------------------------------------------------------------------
__global__ __launch_bounds__(256) void attn_wmma_kernel(
    const __bf16* __restrict__ Qb, const __bf16* __restrict__ Kb,
    const __bf16* __restrict__ Rb, const __bf16* __restrict__ Vtb,
    const float* __restrict__ uvec, const float* __restrict__ vvec,
    __bf16* __restrict__ Ob) {
    extern __shared__ char smem[];
    float*  sc   = (float*)smem;                                  // 64*1032 f32
    __bf16* qu   = (__bf16*)(smem + 64 * SROW * 4);               // 64*72
    __bf16* qv   = qu + 64 * QSTRIDE;                             // 80*72
    float*  rinv = (float*)(qv + 80 * QSTRIDE);                   // 64 f32

    const int qt = blockIdx.x, bh = blockIdx.y;
    const int b = bh >> 4, h = bh & 15;
    const int q0 = qt * 64;
    const int tid = threadIdx.x, wave = tid >> 5, lane = tid & 31;
    const int half = lane >> 4, nn = lane & 15;

    // ---- Stage 1: stage Q tile (rows q0..q0+79) as bf16 (q+u) and (q+v) ----
    const __bf16* Qbase = Qb + (size_t)b * S_LEN * DMODEL + h * DH;
    for (int idx = tid; idx < 80 * DH; idx += 256) {
        const int row = idx >> 6, d = idx & 63;
        int srow = q0 + row; if (srow > S_LEN - 1) srow = S_LEN - 1;
        const float qf = (float)Qbase[srow * DMODEL + d];
        qv[row * QSTRIDE + d] = (__bf16)(qf + vvec[d]);
        if (row < 64) qu[row * QSTRIDE + d] = (__bf16)(qf + uvec[d]);
    }
    __syncthreads();

    // ---- Stage 2: content-content scores ac = (q+u) K^T ----
    const __bf16* Kbase = Kb + (size_t)b * S_LEN * DMODEL + h * DH;
    for (int t = wave; t < 256; t += 8) {          // 4 rowblk x 64 colblk
        const int m0 = (t >> 6) * 16, n0 = (t & 63) * 16;
        v8f acc = {};
#pragma unroll
        for (int kk = 0; kk < DH; kk += 32) {
            v16bf a  = load_frag_a(qu, m0, QSTRIDE, kk, lane);
            v16bf bb = load_frag_b(Kbase, n0, DMODEL, kk, lane);
            acc = wmma_bf16f32(a, bb, acc);
        }
#pragma unroll
        for (int r = 0; r < 8; ++r)
            sc[(m0 + r + half * 8) * SROW + n0 + nn] = acc[r];
    }
    __syncthreads();

    // ---- Stage 3: raw bd = (q+v) R^T over 80 rows, rel-shift scatter-add ----
    // g[j,l] -> sc[j, l+j+1-S] if l >= S-1-j ; else sc[j-1, l+j+1].
    const __bf16* Rbase = Rb + (size_t)b * S_LEN * DMODEL + h * DH;
    for (int t = wave; t < 320; t += 8) {          // 5 rowblk x 64 colblk
        const int m0 = (t >> 6) * 16, l0 = (t & 63) * 16;
        v8f acc = {};
#pragma unroll
        for (int kk = 0; kk < DH; kk += 32) {
            v16bf a  = load_frag_a(qv, m0, QSTRIDE, kk, lane);
            v16bf bb = load_frag_b(Rbase, l0, DMODEL, kk, lane);
            acc = wmma_bf16f32(a, bb, acc);
        }
        const int l = l0 + nn;
#pragma unroll
        for (int r = 0; r < 8; ++r) {
            const int jr = q0 + m0 + r + half * 8;
            if (l >= S_LEN - 1 - jr) {
                const int tr = jr - q0;
                if (tr < 64 && jr < S_LEN)
                    sc[tr * SROW + (l + jr + 1 - S_LEN)] += acc[r];
            } else {
                const int tr = jr - 1 - q0;
                if (tr >= 0 && tr < 64)
                    sc[tr * SROW + (l + jr + 1)] += acc[r];
            }
        }
    }
    __syncthreads();

    // ---- Stage 4: row softmax with wave32 shuffle reductions ----
    for (int rr = 0; rr < 8; ++rr) {
        const int row = wave * 8 + rr;
        float* rp = sc + row * SROW;
        float mx = -1e30f;
        for (int i = lane; i < S_LEN; i += 32) mx = fmaxf(mx, rp[i]);
#pragma unroll
        for (int off = 16; off; off >>= 1) mx = fmaxf(mx, __shfl_xor(mx, off, 32));
        float sum = 0.f;
        for (int i = lane; i < S_LEN; i += 32) {
            const float e = __expf((rp[i] - mx) * ATT_SCALE);
            rp[i] = e;
            sum += e;
        }
#pragma unroll
        for (int off = 16; off; off >>= 1) sum += __shfl_xor(sum, off, 32);
        if (lane == 0) rinv[row] = 1.0f / sum;
    }
    __syncthreads();

    // ---- Stage 5: O = softmax(sc) @ V  (V stored transposed [bh][dh][s]) ----
    const __bf16* Vbase = Vtb + (size_t)bh * DH * S_LEN;
    for (int t = wave * 2; t < wave * 2 + 2; ++t) {   // 4 rowblk x 4 colblk
        const int m0 = (t >> 2) * 16, n0 = (t & 3) * 16;
        const float rs = rinv[m0 + nn];
        v8f acc = {};
        for (int kk = 0; kk < S_LEN; kk += 32) {
            v16bf a  = load_frag_a_probs(sc, m0, SROW, kk, rs, lane);
            v16bf bb = load_frag_b(Vbase, n0, S_LEN, kk, lane);
            acc = wmma_bf16f32(a, bb, acc);
        }
#pragma unroll
        for (int r = 0; r < 8; ++r) {
            const int row = q0 + m0 + r + half * 8;
            Ob[(size_t)(b * S_LEN + row) * DMODEL + h * DH + n0 + nn] =
                (__bf16)acc[r];
        }
    }
}

// ---------------------------------------------------------------------------
// Launcher
// ---------------------------------------------------------------------------
extern "C" void kernel_launch(void* const* d_in, const int* in_sizes, int n_in,
                              void* d_out, int out_size, void* d_ws, size_t ws_size,
                              hipStream_t stream) {
    const float* x   = (const float*)d_in[0];
    const float* u   = (const float*)d_in[1];
    const float* v   = (const float*)d_in[2];
    const float* pos = (const float*)d_in[3];
    const float* Wq  = (const float*)d_in[4];
    const float* bq  = (const float*)d_in[5];
    const float* Wk  = (const float*)d_in[6];
    const float* bk  = (const float*)d_in[7];
    const float* Wv  = (const float*)d_in[8];
    const float* bv  = (const float*)d_in[9];
    const float* Wc  = (const float*)d_in[10];
    const float* bc  = (const float*)d_in[11];

    const size_t XN = (size_t)NTOK * DMODEL;   // 4,194,304
    const size_t WN = (size_t)DMODEL * DMODEL; // 1,048,576

    char* ws = (char*)d_ws;
    __bf16* xb   = (__bf16*)(ws);
    __bf16* posb = xb   + XN;
    __bf16* Wqb  = posb + XN;
    __bf16* Wkb  = Wqb  + WN;
    __bf16* Wvb  = Wkb  + WN;
    __bf16* Wcb  = Wvb  + WN;
    __bf16* Qb   = Wcb  + WN;
    __bf16* Kb   = Qb   + XN;
    __bf16* Vtb  = Kb   + XN;
    __bf16* Ob   = Vtb  + XN;

    // Stage 0: bf16 conversions
    cvt_bf16_kernel<<<(int)(XN / 4 / 256), 256, 0, stream>>>(x,   xb,   (int)(XN / 4));
    cvt_bf16_kernel<<<(int)(XN / 4 / 256), 256, 0, stream>>>(pos, posb, (int)(XN / 4));
    cvt_bf16_kernel<<<(int)(WN / 4 / 256), 256, 0, stream>>>(Wq,  Wqb,  (int)(WN / 4));
    cvt_bf16_kernel<<<(int)(WN / 4 / 256), 256, 0, stream>>>(Wk,  Wkb,  (int)(WN / 4));
    cvt_bf16_kernel<<<(int)(WN / 4 / 256), 256, 0, stream>>>(Wv,  Wvb,  (int)(WN / 4));
    cvt_bf16_kernel<<<(int)(WN / 4 / 256), 256, 0, stream>>>(Wc,  Wcb,  (int)(WN / 4));

    // Stage 1: Q/K/V projections (WMMA GEMMs)
    dim3 ggrid(DMODEL / 64, NTOK / 64);
    gemm_wmma_kernel<<<ggrid, 256, 0, stream>>>(xb, Wqb, bq, Qb,  nullptr, 0);
    gemm_wmma_kernel<<<ggrid, 256, 0, stream>>>(xb, Wkb, bk, Kb,  nullptr, 0);
    gemm_wmma_kernel<<<ggrid, 256, 0, stream>>>(xb, Wvb, bv, Vtb, nullptr, 1);

    // Stage 2: fused relative attention (scores live in LDS)
    const int smem_bytes = 64 * SROW * 4 + (64 + 80) * QSTRIDE * 2 + 64 * 4; // 285,184
    dim3 agrid(S_LEN / 64, BHEADS);
    attn_wmma_kernel<<<agrid, 256, smem_bytes, stream>>>(Qb, Kb, posb, Vtb, u, v, Ob);

    // Stage 3: output projection (f32 out)
    gemm_wmma_kernel<<<ggrid, 256, 0, stream>>>(Ob, Wcb, bc, nullptr, (float*)d_out, 2);
}
